// spatial_illumination_attention_73985106641401
// MI455X (gfx1250) — matrix operand order repaired and checked by reading, not compiled, not run on hardware
//
#include <hip/hip_runtime.h>
#include <hip/hip_bf16.h>

// ---------------------------------------------------------------------------
// spatial_illumination_attention on MI455X (gfx1250, wave32, WMMA)
//   x: (4,64,64,64) f32, W: (64,64) f32, b: (64,) f32
//   e = W@x+b ; S = e^T e ; attn = softmax_j(S) ; out = e @ attn + x
//                                   (contraction over i, softmax over j)
// Fused flash-style: S never hits memory. Shift m_i = ||e_i||*maxnorm
// (Cauchy-Schwarz upper bound -> softmax-invariant, overflow-safe).
// ---------------------------------------------------------------------------

typedef _Float16 v16h __attribute__((ext_vector_type(16)));
typedef _Float16 v8h  __attribute__((ext_vector_type(8)));
typedef float    v8f  __attribute__((ext_vector_type(8)));

#define HW 4096
#define CH 64
#define NB 4
#define LOG2E 1.4426950408889634f

union V16H { v16h v; v8h h[2]; };

// A-fragment (16x32 f16, MxK): lane L holds row M=L&15;
// halfwords 0..15 = K kb..kb+15, kb=(L>>4)*16.
// `p` points at element (row, kb): two contiguous 16B chunks at +0 and +8.
__device__ __forceinline__ v16h frag_A(const _Float16* p) {
  V16H u;
  u.h[0] = *(const v8h*)(p);
  u.h[1] = *(const v8h*)(p + 8);
  return u.v;
}

// B-fragment (32x16 f16, KxN): lane L holds col N=L&15;
// halfwords 0..15 = K kb16..kb16+15, kb16=(L>>4)*16 -> 32B contiguous.
__device__ __forceinline__ v16h frag_B(const _Float16* p) {
  V16H u;
  u.h[0] = *(const v8h*)(p);
  u.h[1] = *(const v8h*)(p + 8);
  return u.v;
}

__device__ __forceinline__ v8f wmma_f16(v16h a, v16h b, v8f c) {
  return __builtin_amdgcn_wmma_f32_16x16x32_f16(
      /*neg_a=*/false, a, /*neg_b=*/false, b,
      /*c_mod=*/(short)0, c, /*reuse_a=*/false, /*reuse_b=*/false);
}

// ---------------------------------------------------------------------------
// Kernel 1: 1x1 conv. Per block: one batch, 64 spatial positions.
// Writes ET[b][i][c] (f16), Ecm[b][c][i] (f16), norms[b][i] = ||e_:,i||.
// ---------------------------------------------------------------------------
__global__ __launch_bounds__(256)
void k_conv(const float* __restrict__ x, const float* __restrict__ W,
            const float* __restrict__ bias,
            _Float16* __restrict__ ET, _Float16* __restrict__ Ecm,
            float* __restrict__ norms) {
  __shared__ float ldsW[CH * CH];
  __shared__ float ldsX[CH * 64];
  __shared__ float normbuf[64];

  const int t  = threadIdx.x;
  const int b  = blockIdx.y;
  const int i0 = blockIdx.x * 64;

  for (int k = t; k < CH * CH; k += 256) ldsW[k] = W[k];
  for (int k = t; k < CH * 64; k += 256) {
    int c = k >> 6, il = k & 63;
    ldsX[c * 64 + il] = x[((size_t)(b * CH + c) << 12) + i0 + il];
  }
  if (t < 64) normbuf[t] = 0.0f;
  __syncthreads();

  const int il = t & 63;     // spatial position within tile
  const int og = t >> 6;     // output-channel group (16 channels each)

  float acc[16];
#pragma unroll
  for (int oo = 0; oo < 16; ++oo) acc[oo] = 0.0f;
  for (int c = 0; c < CH; ++c) {
    float xv = ldsX[c * 64 + il];
#pragma unroll
    for (int oo = 0; oo < 16; ++oo)
      acc[oo] = fmaf(ldsW[(og * 16 + oo) * CH + c], xv, acc[oo]);
  }

  float nrm2 = 0.0f;
  _Float16 eh[16];
#pragma unroll
  for (int oo = 0; oo < 16; ++oo) {
    int o = og * 16 + oo;
    float e = acc[oo] + bias[o];
    nrm2 += e * e;
    eh[oo] = (_Float16)e;
    Ecm[((size_t)b * CH + o) * HW + i0 + il] = eh[oo];
  }
  // ET row (c-contiguous): 16 halfs = 32B contiguous
  {
    _Float16* dst = ET + ((size_t)(b * HW + i0 + il)) * CH + og * 16;
    *(v8h*)(dst)     = *(v8h*)(&eh[0]);
    *(v8h*)(dst + 8) = *(v8h*)(&eh[8]);
  }
  atomicAdd(&normbuf[il], nrm2);
  __syncthreads();
  if (t < 64) norms[b * HW + i0 + t] = sqrtf(normbuf[t]);
}

// ---------------------------------------------------------------------------
// Kernel 2: per-batch max of column norms (+ slack for f16 rounding).
// ---------------------------------------------------------------------------
__global__ __launch_bounds__(256)
void k_maxnorm(const float* __restrict__ norms, float* __restrict__ maxn) {
  __shared__ float red[256];
  const int b = blockIdx.x;
  float m = 0.0f;
  for (int k = threadIdx.x; k < HW; k += 256) m = fmaxf(m, norms[b * HW + k]);
  red[threadIdx.x] = m;
  __syncthreads();
  for (int s = 128; s > 0; s >>= 1) {
    if (threadIdx.x < s) red[threadIdx.x] = fmaxf(red[threadIdx.x], red[threadIdx.x + s]);
    __syncthreads();
  }
  if (threadIdx.x == 0) maxn[b] = red[0] + 1e-2f;
}

// ---------------------------------------------------------------------------
// Kernel 3: row log-sum-exp. One wave per 16-row i-tile, sweeps all j-tiles.
//   q[i] = mk_i + log2( sum_j exp2(LOG2E*S[i,j] - mk_i) ),  mk_i = k*||e_i||*maxnorm
// ---------------------------------------------------------------------------
__global__ __launch_bounds__(128)
void k_rowsum(const _Float16* __restrict__ ET, const float* __restrict__ norms,
              const float* __restrict__ maxn, float* __restrict__ qbuf) {
  const int lane = threadIdx.x & 31;
  const int wv   = threadIdx.x >> 5;
  const int b    = blockIdx.y;
  const int i0   = (blockIdx.x * 4 + wv) * 16;

  const _Float16* ETb = ET + (size_t)b * HW * CH;
  const int r  = lane & 15;
  const int hi = lane >> 4;
  const int kb = hi * 16;

  // A fragments: rows i0..i0+15 of E^T, K-halves c=0..31 and c=32..63
  const _Float16* arow = ETb + (size_t)(i0 + r) * CH;
  const v16h A0 = frag_A(arow + kb);
  const v16h A1 = frag_A(arow + 32 + kb);

  const float mb = maxn[b];
  float mkv[8];
#pragma unroll
  for (int v = 0; v < 8; ++v)
    mkv[v] = norms[b * HW + i0 + v + 8 * hi] * mb * LOG2E;

  float acc[8];
#pragma unroll
  for (int v = 0; v < 8; ++v) acc[v] = 0.0f;

  const int kb16 = hi * 16;
  for (int jt = 0; jt < 256; ++jt) {
    const _Float16* brow = ETb + (size_t)(jt * 16 + r) * CH + kb16;
    v16h B0 = frag_B(brow);        // K = c 0..31 half
    v16h B1 = frag_B(brow + 32);   // K = c 32..63 half
    v8f s = {};
    s = wmma_f16(A0, B0, s);
    s = wmma_f16(A1, B1, s);
#pragma unroll
    for (int v = 0; v < 8; ++v)
      acc[v] += __builtin_amdgcn_exp2f(s[v] * LOG2E - mkv[v]);
  }
  // reduce over the 16 lanes of each half (rows v / v+8 live per-half)
#pragma unroll
  for (int v = 0; v < 8; ++v) {
    acc[v] += __shfl_xor(acc[v], 1, 32);
    acc[v] += __shfl_xor(acc[v], 2, 32);
    acc[v] += __shfl_xor(acc[v], 4, 32);
    acc[v] += __shfl_xor(acc[v], 8, 32);
  }
  if (lane == 0) {
#pragma unroll
    for (int v = 0; v < 8; ++v)
      qbuf[b * HW + i0 + v] = mkv[v] + __builtin_amdgcn_logf(acc[v]);
  }
  if (lane == 16) {
#pragma unroll
    for (int v = 0; v < 8; ++v)
      qbuf[b * HW + i0 + 8 + v] = mkv[v] + __builtin_amdgcn_logf(acc[v]);
  }
}

// ---------------------------------------------------------------------------
// Kernel 4: fused attention. One wave per 16-col j-tile; sweep i in chunks
// of 32: S-tiles (WMMA) -> exp2(k*S - q_i) -> f16 via LDS -> B-fragment ->
// accumulate F[64x16] with 4 WMMAs. Epilogue adds residual x.
// ---------------------------------------------------------------------------
__global__ __launch_bounds__(128)
void k_attn(const _Float16* __restrict__ ET, const _Float16* __restrict__ Ecm,
            const float* __restrict__ qbuf, const float* __restrict__ x,
            float* __restrict__ out) {
  // per-wave P'' staging: 16 cols x 32 rows, stride 40 halfs (80B) kills
  // bank conflicts (gcd trick) and keeps 16B alignment for ds_load_b128.
  __shared__ __align__(16) _Float16 pp[4][16 * 40];

  const int lane = threadIdx.x & 31;
  const int wv   = threadIdx.x >> 5;
  const int b    = blockIdx.y;
  const int j0   = (blockIdx.x * 4 + wv) * 16;

  const _Float16* ETb = ET  + (size_t)b * HW * CH;
  const _Float16* Ecb = Ecm + (size_t)b * CH * HW;
  const float*    qb  = qbuf + b * HW;

  const int n    = lane & 15;
  const int hi   = lane >> 4;
  const int kb   = hi * 16;
  const int kb16 = hi * 16;

  // S-GEMM B fragments are fixed for this wave (columns j0..j0+15)
  const _Float16* brow = ETb + (size_t)(j0 + n) * CH + kb16;
  const v16h Bs0 = frag_B(brow);
  const v16h Bs1 = frag_B(brow + 32);

  v8f facc[4] = {{}, {}, {}, {}};
  _Float16* myp = pp[wv];

  for (int kc = 0; kc < 128; ++kc) {
    const int i0 = kc * 32;

    // S tiles: rows i0..i0+15 and i0+16..i0+31
    const _Float16* arow0 = ETb + (size_t)(i0 + n) * CH;
    const _Float16* arow1 = arow0 + 16 * CH;
    __builtin_prefetch(arow0 + 32 * CH, 0, 1);  // next chunk's rows
    v16h As00 = frag_A(arow0 + kb),      As01 = frag_A(arow0 + 32 + kb);
    v16h As10 = frag_A(arow1 + kb),      As11 = frag_A(arow1 + 32 + kb);

    v8f s0 = {}, s1 = {};
    s0 = wmma_f16(As00, Bs0, s0);
    s0 = wmma_f16(As01, Bs1, s0);
    s1 = wmma_f16(As10, Bs0, s1);
    s1 = wmma_f16(As11, Bs1, s1);

    // attn probs: exp2(LOG2E*S - q_i)  (== exp(S-m_i)/l_i, in [0,1])
#pragma unroll
    for (int v = 0; v < 8; ++v) {
      const int rv = v + 8 * hi;
      float p0 = __builtin_amdgcn_exp2f(s0[v] * LOG2E - qb[i0 + rv]);
      float p1 = __builtin_amdgcn_exp2f(s1[v] * LOG2E - qb[i0 + 16 + rv]);
      myp[n * 40 + rv]      = (_Float16)p0;
      myp[n * 40 + 16 + rv] = (_Float16)p1;
    }
    // same wave produced & consumes this LDS region: only need DScnt drain
    asm volatile("s_wait_dscnt 0" ::: "memory");

    const v16h Bp = frag_B(myp + n * 40 + kb16);

    // F += E(c-tile, i-chunk) * P''
#pragma unroll
    for (int m = 0; m < 4; ++m) {
      const _Float16* frow = Ecb + (size_t)(m * 16 + n) * HW + i0 + kb;
      v16h Af = frag_A(frow);
      facc[m] = wmma_f16(Af, Bp, facc[m]);
    }
  }

  // epilogue: out = F + x   (D layout: VGPR v, half hi -> row m*16+v+8*hi)
#pragma unroll
  for (int m = 0; m < 4; ++m) {
#pragma unroll
    for (int v = 0; v < 8; ++v) {
      const int c = m * 16 + v + 8 * hi;
      const size_t idx = ((size_t)(b * CH + c)) * HW + j0 + n;
      out[idx] = facc[m][v] + x[idx];
    }
  }
}

// ---------------------------------------------------------------------------
extern "C" void kernel_launch(void* const* d_in, const int* in_sizes, int n_in,
                              void* d_out, int out_size, void* d_ws, size_t ws_size,
                              hipStream_t stream) {
  const float* x    = (const float*)d_in[0];
  const float* W    = (const float*)d_in[1];
  const float* bias = (const float*)d_in[2];
  float* out = (float*)d_out;

  char* ws = (char*)d_ws;
  _Float16* ET  = (_Float16*)ws;                                   // 2 MiB
  _Float16* Ecm = (_Float16*)(ws + (size_t)NB * HW * CH * 2);      // 2 MiB
  float* norms  = (float*)(ws + (size_t)2 * NB * HW * CH * 2);     // 64 KiB
  float* qbuf   = norms + NB * HW;                                 // 64 KiB
  float* maxn   = qbuf + NB * HW;                                  // 16 B

  k_conv<<<dim3(64, NB), 256, 0, stream>>>(x, W, bias, ET, Ecm, norms);
  k_maxnorm<<<NB, 256, 0, stream>>>(norms, maxn);
  k_rowsum<<<dim3(64, NB), 128, 0, stream>>>(ET, norms, maxn, qbuf);
  k_attn<<<dim3(64, NB), 128, 0, stream>>>(ET, Ecm, qbuf, x, out);
}
